// LSTM_3865470566695
// MI455X (gfx1250) — compile-verified, matrix operand order checked by reading, not run on hardware
//
#include <hip/hip_runtime.h>

// LSTM on MI455X (gfx1250).
// Decomposition: 8 batch tiles (WMMA M=16) x 4 hidden-column groups = 32 WGs,
// 256 threads (8 waves) each. Wave w of group g owns hidden columns
// [g*128 + w*16, +16) and keeps i/f/o/g accumulators and the cell state c in
// registers for all 2048 steps. h is exchanged among the 4 WGs of a batch tile
// through an L2-resident f16 buffer guarded by a monotonic atomic barrier
// (2 phases/step: publish, then copy-back) -- no grid-wide sync, no resets.
// Weights are pre-converted to f16 in d_ws (WMMA f32_16x16x32_f16 path; the
// f32 WMMA only has K=4 and is ~8x slower per instruction).
// ws_size needed: ~2.9 MB.

typedef _Float16 half_t;
typedef __attribute__((ext_vector_type(16))) _Float16 v16h;
typedef __attribute__((ext_vector_type(8)))  _Float16 v8h;
typedef __attribute__((ext_vector_type(8)))  float    v8f;
typedef __attribute__((ext_vector_type(4)))  float    v4f;

#define S_LEN  2048
#define B_SZ   128
#define I_SZ   128
#define H_SZ   512
#define O_SZ   128
#define NGROUP 4            // hidden-column groups (WGs per batch tile)

#define H_STRIDE 520        // 512 + 8 halves pad: 1040 B rows (16B aligned)
#define X_STRIDE 136        // 128 + 8 halves pad: 272 B rows (16B aligned)

__device__ __forceinline__ float sigmoidf_fast(float x) {
    return 1.0f / (1.0f + __expf(-x));
}

// ISA 7.12.2 16-bit A/B fragment for 16x16x32: per lane, VGPRs 0..3 hold
// K = base..base+7 and VGPRs 4..7 hold K = base+16..base+23, base = kk+8*(lane>>4).
__device__ __forceinline__ v16h load_frag(const half_t* __restrict__ p) {
    v8h lo = *(const v8h*)(p);
    v8h hi = *(const v8h*)(p + 16);
    v16h r;
#pragma unroll
    for (int i = 0; i < 8; ++i) { r[i] = lo[i]; r[i + 8] = hi[i]; }
    return r;
}

__device__ __forceinline__ v8f wmma_f16(v16h a, v16h b, v8f c) {
    return __builtin_amdgcn_wmma_f32_16x16x32_f16(false, a, false, b, (short)0, c,
                                                  false, false);
}

__global__ void f32_to_f16_kernel(const float* __restrict__ src,
                                  half_t* __restrict__ dst, int n) {
    int i = blockIdx.x * blockDim.x + threadIdx.x;
    if (i < n) dst[i] = (half_t)src[i];
}

__global__ void init_ctr_kernel(unsigned int* __restrict__ ctr) {
    if (threadIdx.x < 8) ctr[threadIdx.x] = 0u;
}

__global__ __launch_bounds__(256) void lstm_persistent_kernel(
    const float* __restrict__ x,       // [B, S, I] f32
    const half_t* __restrict__ Wh,     // [4H, H] f16
    const half_t* __restrict__ Wx,     // [4H, I] f16
    const float* __restrict__ b,       // [4H]    f32
    const half_t* __restrict__ Wo,     // [O, H]  f16
    const float* __restrict__ bo,      // [O]     f32
    half_t* __restrict__ hbuf,         // [8][16][512] f16 h-exchange (d_ws)
    unsigned int* __restrict__ ctr,    // [8] monotonic barrier counters (d_ws)
    float* __restrict__ out)           // [B, O]  f32
{
    __shared__ __align__(16) half_t hlds[16 * H_STRIDE]; // full h tile [m][k]
    __shared__ __align__(16) half_t xlds[16 * X_STRIDE]; // x_t tile [m][k]

    const int tid    = threadIdx.x;     // 0..255
    const int wave   = tid >> 5;        // 0..7
    const int lane   = tid & 31;
    const int nlane  = lane & 15;
    const int halfid = lane >> 4;
    const int bt     = blockIdx.x;      // batch tile 0..7
    const int grp    = blockIdx.y;      // hidden group 0..3
    const int hcol   = grp * 128 + wave * 16 + nlane; // owned hidden column
    const int m      = nlane;                          // A-matrix row

    const float bias_i = b[0 * H_SZ + hcol];
    const float bias_f = b[1 * H_SZ + hcol];
    const float bias_o = b[2 * H_SZ + hcol];
    const float bias_g = b[3 * H_SZ + hcol];

    for (int i = tid; i < 16 * H_STRIDE; i += 256) hlds[i] = (half_t)0.0f;

    float cc[8];
#pragma unroll
    for (int v = 0; v < 8; ++v) cc[v] = 0.0f;

    // copy-back geometry: thread copies 32 halves of one h row from hbuf
    const int cr = tid >> 4;            // row 0..15
    const int ccol = (tid & 15) * 32;   // col 0..480

    __syncthreads();

    for (int t = 0; t < S_LEN; ++t) {
        // Stage x_t (16x128) into LDS as f16; nontemporal: x is streamed once
        // and must not evict the L2-resident weights.
        {
            int r  = tid >> 4;                  // 0..15
            int i0 = (tid & 15) * 8;            // 0..120
            const float* src = x + ((size_t)(bt * 16 + r) * S_LEN + t) * I_SZ + i0;
            v8h h8;
#pragma unroll
            for (int j = 0; j < 8; ++j) h8[j] = (half_t)__builtin_nontemporal_load(src + j);
            *(v8h*)&xlds[r * X_STRIDE + i0] = h8;
        }
        __syncthreads();   // xlds staged, hlds copy from previous step visible

        v8f acc0, acc1, acc2, acc3;
#pragma unroll
        for (int v = 0; v < 8; ++v) {
            acc0[v] = bias_i; acc1[v] = bias_f; acc2[v] = bias_o; acc3[v] = bias_g;
        }

        // Recurrent term: K = 512 over h (A from LDS, B rows of Wh from L2).
#pragma unroll 2
        for (int kk = 0; kk < H_SZ; kk += 32) {
            v16h a  = load_frag(&hlds[m * H_STRIDE + kk + halfid * 8]);
            v16h b0 = load_frag(&Wh[(size_t)(0 * H_SZ + hcol) * H_SZ + kk + halfid * 8]);
            v16h b1 = load_frag(&Wh[(size_t)(1 * H_SZ + hcol) * H_SZ + kk + halfid * 8]);
            v16h b2 = load_frag(&Wh[(size_t)(2 * H_SZ + hcol) * H_SZ + kk + halfid * 8]);
            v16h b3 = load_frag(&Wh[(size_t)(3 * H_SZ + hcol) * H_SZ + kk + halfid * 8]);
            acc0 = wmma_f16(a, b0, acc0);
            acc1 = wmma_f16(a, b1, acc1);
            acc2 = wmma_f16(a, b2, acc2);
            acc3 = wmma_f16(a, b3, acc3);
        }
        // Input term: K = 128 over x_t.
#pragma unroll 2
        for (int kk = 0; kk < I_SZ; kk += 32) {
            v16h a  = load_frag(&xlds[m * X_STRIDE + kk + halfid * 8]);
            v16h b0 = load_frag(&Wx[(size_t)(0 * H_SZ + hcol) * I_SZ + kk + halfid * 8]);
            v16h b1 = load_frag(&Wx[(size_t)(1 * H_SZ + hcol) * I_SZ + kk + halfid * 8]);
            v16h b2 = load_frag(&Wx[(size_t)(2 * H_SZ + hcol) * I_SZ + kk + halfid * 8]);
            v16h b3 = load_frag(&Wx[(size_t)(3 * H_SZ + hcol) * I_SZ + kk + halfid * 8]);
            acc0 = wmma_f16(a, b0, acc0);
            acc1 = wmma_f16(a, b1, acc1);
            acc2 = wmma_f16(a, b2, acc2);
            acc3 = wmma_f16(a, b3, acc3);
        }

        // Gate nonlinearities + cell update (register-local: this wave holds all
        // four gates of its 16 hidden columns).
#pragma unroll
        for (int v = 0; v < 8; ++v) {
            float iv = sigmoidf_fast(acc0[v]);
            float fv = sigmoidf_fast(acc1[v]);
            float ov = sigmoidf_fast(acc2[v]);
            float gv = tanhf(acc3[v]);
            float cn = fv * cc[v] + iv * gv;
            cc[v] = cn;
            // Publish owned h columns to the exchange buffer (L2-resident).
            hbuf[((size_t)bt * 16 + (v + 8 * halfid)) * H_SZ + hcol] =
                (half_t)(ov * tanhf(cn));
        }
        __threadfence();       // release h stores (agent scope)
        __syncthreads();       // also: all matmul reads of hlds are done

        // Phase 1: all 4 groups published h for step t.
        if (tid == 0) {
            __hip_atomic_fetch_add(ctr + bt, 1u, __ATOMIC_RELEASE,
                                   __HIP_MEMORY_SCOPE_AGENT);
            unsigned int tgt = 2u * NGROUP * (unsigned)t + NGROUP;
            while (__hip_atomic_load(ctr + bt, __ATOMIC_ACQUIRE,
                                     __HIP_MEMORY_SCOPE_AGENT) < tgt)
                __builtin_amdgcn_s_sleep(2);
        }
        __syncthreads();
        __threadfence();       // acquire: drop stale hbuf lines from WGP$

        // Copy the full h (16x512) from hbuf into LDS (32 halves / thread).
        {
            const half_t* src = hbuf + ((size_t)bt * 16 + cr) * H_SZ + ccol;
            half_t* dst = &hlds[cr * H_STRIDE + ccol];
#pragma unroll
            for (int j = 0; j < 4; ++j)
                ((v8h*)dst)[j] = ((const v8h*)src)[j];
        }
        __syncthreads();       // copy reads of hbuf complete

        // Phase 2: all 4 groups finished copying -> safe to overwrite hbuf at t+1.
        if (tid == 0) {
            __hip_atomic_fetch_add(ctr + bt, 1u, __ATOMIC_RELEASE,
                                   __HIP_MEMORY_SCOPE_AGENT);
            unsigned int tgt = 2u * NGROUP * (unsigned)(t + 1);
            while (__hip_atomic_load(ctr + bt, __ATOMIC_ACQUIRE,
                                     __HIP_MEMORY_SCOPE_AGENT) < tgt)
                __builtin_amdgcn_s_sleep(2);
        }
        __syncthreads();
    }

    // Output projection: out[16,128] = h_last @ Wo^T + bo. The 8 output tiles
    // are split 2-per-group; wave-uniform branch keeps EXEC all-ones for WMMA.
    if (wave < 2) {
        const int ocol = (grp * 2 + wave) * 16 + nlane;
        const float bov = bo[ocol];
        v8f acc;
#pragma unroll
        for (int v = 0; v < 8; ++v) acc[v] = bov;
#pragma unroll 2
        for (int kk = 0; kk < H_SZ; kk += 32) {
            v16h a  = load_frag(&hlds[m * H_STRIDE + kk + halfid * 8]);
            v16h bw = load_frag(&Wo[(size_t)ocol * H_SZ + kk + halfid * 8]);
            acc = wmma_f16(a, bw, acc);
        }
#pragma unroll
        for (int v = 0; v < 8; ++v) {
            int row = bt * 16 + v + 8 * halfid;
            out[(size_t)row * O_SZ + ocol] = acc[v];
        }
    }
}

extern "C" void kernel_launch(void* const* d_in, const int* in_sizes, int n_in,
                              void* d_out, int out_size, void* d_ws, size_t ws_size,
                              hipStream_t stream) {
    const float* x  = (const float*)d_in[0];   // [128, 2048, 128]
    const float* Wh = (const float*)d_in[1];   // [2048, 512]
    const float* Wx = (const float*)d_in[2];   // [2048, 128]
    const float* b  = (const float*)d_in[3];   // [1, 2048]
    const float* Wo = (const float*)d_in[4];   // [128, 512]
    const float* bo = (const float*)d_in[5];   // [128]
    float* out = (float*)d_out;                // [128, 128]

    const int nWh = 4 * H_SZ * H_SZ;   // 1,048,576
    const int nWx = 4 * H_SZ * I_SZ;   //   262,144
    const int nWo = O_SZ * H_SZ;       //    65,536
    const int nHb = 8 * 16 * H_SZ;     //    65,536 (h exchange)

    half_t* Wh_h = (half_t*)d_ws;
    half_t* Wx_h = Wh_h + nWh;
    half_t* Wo_h = Wx_h + nWx;
    half_t* hbuf = Wo_h + nWo;
    unsigned int* ctr = (unsigned int*)(hbuf + nHb);   // 8 uints, 4B aligned

    f32_to_f16_kernel<<<(nWh + 255) / 256, 256, 0, stream>>>(Wh, Wh_h, nWh);
    f32_to_f16_kernel<<<(nWx + 255) / 256, 256, 0, stream>>>(Wx, Wx_h, nWx);
    f32_to_f16_kernel<<<(nWo + 255) / 256, 256, 0, stream>>>(Wo, Wo_h, nWo);
    init_ctr_kernel<<<1, 32, 0, stream>>>(ctr);

    lstm_persistent_kernel<<<dim3(B_SZ / 16, NGROUP), 256, 0, stream>>>(
        x, Wh_h, Wx_h, b, Wo_h, bo, hbuf, ctr, out);
}
